// EALSTM_20495583936932
// MI455X (gfx1250) — matrix operand order repaired and checked by reading, not compiled, not run on hardware
//
#include <hip/hip_runtime.h>

typedef __attribute__((ext_vector_type(16))) _Float16 v16h;
typedef __attribute__((ext_vector_type(8)))  _Float16 v8h;
typedef __attribute__((ext_vector_type(8)))  float    v8f;

#define B_    256
#define T_    365
#define DDYN  32
#define DSTAT 5
#define H_    256
#define G3    768

// Workspace layout (f16 halfs), pre-swizzled WMMA B-fragments:
//   hh fragment (w,g,kc): offset ((w*3+g)*8 + kc)*512 + lane*16 + i
//      holds W_hh[kc*32 + (lane>>4)*8 + (i&7) + (i>>3)*16][g*256 + w*16 + (lane&15)]
//   ih fragment (w,g):    offset HH_HALFS + (w*3+g)*512 + lane*16 + i
//      holds W_ih[(lane>>4)*8 + (i&7) + (i>>3)*16][g*256 + w*16 + (lane&15)]
#define HH_HALFS (16 * 3 * 8 * 512)   // 196608 halfs = 384 KB
#define IH_HALFS (16 * 3 * 512)       //  24576 halfs =  48 KB

#define LDS_STRIDE 264                 // halfs/row: 528B stride -> 64-bank conflict-free
#define LDS_BUF    (16 * LDS_STRIDE)   // one h buffer

__global__ __launch_bounds__(256) void ealstm_prep(const float* __restrict__ Wih,
                                                   const float* __restrict__ Whh,
                                                   _Float16* __restrict__ wsh) {
  int idx = blockIdx.x * blockDim.x + threadIdx.x;
  if (idx >= HH_HALFS + IH_HALFS) return;
  if (idx < HH_HALFS) {
    int i    = idx & 15;
    int lane = (idx >> 4) & 31;
    int frag = idx >> 9;
    int kc = frag & 7;
    int g  = (frag >> 3) % 3;
    int w  = frag / 24;
    int k   = kc * 32 + (lane >> 4) * 8 + (i & 7) + (i >> 3) * 16;
    int col = g * 256 + w * 16 + (lane & 15);
    wsh[idx] = (_Float16)Whh[k * G3 + col];
  } else {
    int r    = idx - HH_HALFS;
    int i    = r & 15;
    int lane = (r >> 4) & 31;
    int frag = r >> 9;
    int g = frag % 3;
    int w = frag / 3;
    int k   = (lane >> 4) * 8 + (i & 7) + (i >> 3) * 16;
    int col = g * 256 + w * 16 + (lane & 15);
    wsh[idx] = (_Float16)Wih[k * G3 + col];
  }
}

__device__ __forceinline__ float sigmoidf_(float x) {
  return 1.0f / (1.0f + __expf(-x));
}
__device__ __forceinline__ float tanhf_(float x) {
  float ax = fabsf(x);
  float e  = __expf(2.0f * ax);
  float t  = 1.0f - 2.0f / (e + 1.0f);
  return copysignf(t, x);
}

// 16 blocks (one 16-row batch tile each) x 512 threads (16 waves; wave w owns
// H-columns [w*16, w*16+16) and its f/o/g gate tiles).
__global__ __launch_bounds__(512) void ealstm_main(
    const float* __restrict__ xd, const float* __restrict__ xs,
    const float* __restrict__ Wsh, const float* __restrict__ bias,
    const float* __restrict__ bias_s, const _Float16* __restrict__ wsh,
    float* __restrict__ out) {
  __shared__ _Float16 h_buf[2 * LDS_BUF];   // double-buffered hidden state (f16)

  const int tid  = threadIdx.x;
  const int w    = tid >> 5;        // wave id = H-tile id
  const int lane = tid & 31;
  const int n    = lane & 15;       // tile column / A-fragment row
  const int lh   = lane >> 4;       // lane half-group
  const int b0   = blockIdx.x * 16; // batch tile base

  // zero initial hidden state (buffer 0 only; buffer 1 is written before read)
  for (int i = tid; i < LDS_BUF; i += 512) h_buf[i] = (_Float16)0.0f;

  const int hcol = w * 16 + n;

  // static input gate block (C/D layout: m = r + 8*lh, col = hcol), registers
  v8f ig;
  #pragma unroll
  for (int r = 0; r < 8; ++r) {
    int b = b0 + r + lh * 8;
    float a = bias_s[hcol];
    #pragma unroll
    for (int d = 0; d < DSTAT; ++d) a += xs[b * DSTAT + d] * Wsh[d * H_ + hcol];
    ig[r] = sigmoidf_(a);
  }

  // gate biases, added in the elementwise phase (C of the first WMMA is inline 0)
  const float bF = bias[0 * 256 + hcol];
  const float bO = bias[1 * 256 + hcol];
  const float bG = bias[2 * 256 + hcol];

  // Register-resident fragments: W_ih (all gates) + W_hh for f and o gates.
  v16h Bih[3];
  #pragma unroll
  for (int g = 0; g < 3; ++g)
    Bih[g] = *(const v16h*)(wsh + HH_HALFS + ((w * 3 + g) << 9) + (lane << 4));
  v16h Bf[8], Bo[8];
  #pragma unroll
  for (int kc = 0; kc < 8; ++kc) {
    Bf[kc] = *(const v16h*)(wsh + (((w * 3 + 0) * 8 + kc) << 9) + (lane << 4));
    Bo[kc] = *(const v16h*)(wsh + (((w * 3 + 1) * 8 + kc) << 9) + (lane << 4));
  }
  // g-gate W_hh fragments are streamed from L2 every step (fits reg budget).
  const _Float16* pg_base = wsh + (size_t)(((w * 3 + 2) * 8) << 9) + (lane << 4);

  // software-pipelined x_d row (16 f32 per lane, A 16x32 fragment order)
  const float* xbase = xd + (size_t)(b0 + n) * T_ * DDYN + lh * 8;
  float xf[16];
  #pragma unroll
  for (int j = 0; j < 8; ++j) { xf[j] = xbase[j]; xf[j + 8] = xbase[j + 16]; }

  v8f c_reg = {};
  float* outh = out;
  float* outc = out + (size_t)B_ * T_ * H_;

  __syncthreads();

  #pragma unroll 1
  for (int t = 0; t < T_; ++t) {
    const _Float16* cur = h_buf + ((t & 1) ? LDS_BUF : 0);
    _Float16*       nxt = h_buf + ((t & 1) ? 0 : LDS_BUF);

    // opaque per-iteration byte offset: blocks LICM from hoisting the g-gate
    // weight loads (keeps them streaming from L2), while preserving the
    // kernarg provenance so they lower to global_load (LOADcnt only).
    int soff = 0;
    asm volatile("" : "+v"(soff));
    const _Float16* pg = pg_base + soff;

    // convert current x fragment, then immediately issue next step's loads
    v16h ax;
    #pragma unroll
    for (int j = 0; j < 16; ++j) ax[j] = (_Float16)xf[j];
    {
      int tn = (t + 1 < T_) ? (t + 1) : t;
      const float* xn = xbase + (size_t)tn * DDYN;
      #pragma unroll
      for (int j = 0; j < 8; ++j) { xf[j] = xn[j]; xf[j + 8] = xn[j + 16]; }
    }

    v8f acc[3];

    // gates = h @ W_hh  (K = 256 as 8 chunks of 32); kc==0 uses inline-0 C
    #pragma unroll
    for (int kc = 0; kc < 8; ++kc) {
      const _Float16* ar = cur + n * LDS_STRIDE + kc * 32 + lh * 8;
      v8h alo = *(const v8h*)(ar);
      v8h ahi = *(const v8h*)(ar + 16);
      v16h ah;
      #pragma unroll
      for (int j = 0; j < 8; ++j) { ah[j] = alo[j]; ah[j + 8] = ahi[j]; }
      v16h bg = *(const v16h*)(pg + (kc << 9));   // streamed each step
      if (kc == 0) {
        v8f z = {};
        acc[0] = __builtin_amdgcn_wmma_f32_16x16x32_f16(false, ah, false, Bf[0],
                                                        (short)0, z, false, false);
        acc[1] = __builtin_amdgcn_wmma_f32_16x16x32_f16(false, ah, false, Bo[0],
                                                        (short)0, z, false, false);
        acc[2] = __builtin_amdgcn_wmma_f32_16x16x32_f16(false, ah, false, bg,
                                                        (short)0, z, false, false);
      } else {
        acc[0] = __builtin_amdgcn_wmma_f32_16x16x32_f16(false, ah, false, Bf[kc],
                                                        (short)0, acc[0], false, false);
        acc[1] = __builtin_amdgcn_wmma_f32_16x16x32_f16(false, ah, false, Bo[kc],
                                                        (short)0, acc[1], false, false);
        acc[2] = __builtin_amdgcn_wmma_f32_16x16x32_f16(false, ah, false, bg,
                                                        (short)0, acc[2], false, false);
      }
    }

    // gates += x_t @ W_ih (last: its operand loads/cvts enjoy the hh shadow)
    #pragma unroll
    for (int g = 0; g < 3; ++g)
      acc[g] = __builtin_amdgcn_wmma_f32_16x16x32_f16(false, ax, false, Bih[g],
                                                      (short)0, acc[g], false, false);

    // elementwise LSTM update; f=acc[0], o=acc[1], g=acc[2]; bias added here
    #pragma unroll
    for (int r = 0; r < 8; ++r) {
      int m = r + lh * 8;
      float fv = sigmoidf_(acc[0][r] + bF);
      float ov = sigmoidf_(acc[1][r] + bO);
      float gv = tanhf_(acc[2][r] + bG);
      float cv = fv * c_reg[r] + ig[r] * gv;
      float hv = ov * tanhf_(cv);
      c_reg[r] = cv;
      nxt[m * LDS_STRIDE + hcol] = (_Float16)hv;
      size_t o = ((size_t)(b0 + m) * T_ + t) * H_ + hcol;
      outh[o] = hv;
      outc[o] = cv;
    }

    // single barrier per step: publishes nxt, and proves all reads of cur
    // are done before it gets overwritten in the next iteration.
    __syncthreads();
  }
}

extern "C" void kernel_launch(void* const* d_in, const int* in_sizes, int n_in,
                              void* d_out, int out_size, void* d_ws, size_t ws_size,
                              hipStream_t stream) {
  const float* xd    = (const float*)d_in[0];  // x_d      [B,T,32]
  const float* xs    = (const float*)d_in[1];  // x_s      [B,5]
  const float* Wih   = (const float*)d_in[2];  // weight_ih[32,768]
  const float* Whh   = (const float*)d_in[3];  // weight_hh[256,768]
  const float* Wsh   = (const float*)d_in[4];  // weight_sh[5,256]
  const float* bias  = (const float*)d_in[5];  // bias     [768]
  const float* biass = (const float*)d_in[6];  // bias_s   [256]
  float* outp   = (float*)d_out;               // [h_n ; c_n] each [B,T,H]
  _Float16* wsh = (_Float16*)d_ws;             // needs 442368 bytes

  const int total = HH_HALFS + IH_HALFS;
  ealstm_prep<<<(total + 255) / 256, 256, 0, stream>>>(Wih, Whh, wsh);
  ealstm_main<<<B_ / 16, 512, 0, stream>>>(xd, xs, Wsh, bias, biass, wsh, outp);
}
